// ARLSTM_67473936220828
// MI455X (gfx1250) — compile-verified, matrix operand order checked by reading, not run on hardware
//
#include <hip/hip_runtime.h>

typedef __attribute__((ext_vector_type(16))) _Float16 v16h;
typedef __attribute__((ext_vector_type(8)))  _Float16 v8h;
typedef __attribute__((ext_vector_type(8)))  float    v8f;

#define HID   256          // hidden size
#define DIN   16           // dynamic input dim
#define EIN   64           // embedding dim
#define SIN   32           // statics dim
#define KEXT  288          // fused K: 256 h | 16 x | 1 Q | 15 pad
#define NKT   9            // K tiles of 32
#define HSTR  296          // state row stride in halves (592B rows -> good bank spread)
#define NW    16           // waves per workgroup
#define BM    16           // batch rows per workgroup
#define WFRAG 1024         // bytes per B-fragment (32 lanes x 32B)
#define LDSKT 4            // K-tiles resident in LDS (kt 0..3)
#define HBUF_BYTES  (BM * HSTR * 2)                 // 9472 (16B aligned), x2 ping-pong
#define WLDS_OFF    (2 * HBUF_BYTES)                // 18944
#define WBLK        (LDSKT * 4 * WFRAG + 4 * 512)   // per-wave block: 16KB frags + 2KB sp
#define SPV_OFF     (LDSKT * 4 * WFRAG)             // sp offset within wave block = 16384
#define LDS_BYTES   (WLDS_OFF + NW * WBLK)          // 18944 + 294912 = 313856 < 320K

__device__ __forceinline__ float sigmoidf_(float x)  { return 1.0f / (1.0f + __expf(-x)); }
__device__ __forceinline__ float tanhf_fast(float x) { return 2.0f / (1.0f + __expf(-2.0f * x)) - 1.0f; }

union AFrag { struct { v8h lo, hi; } p; v16h v; };

// Fused weight matrix W'[K=288][N=1024]:
//   K   0..255 : W_hh[n][K]                                           (recurrent)
//   K 256..271 : (fc? 0.5*W_emb_fc : W_emb_hc)[K-256,:] @ W_ih[n,:64] (input path)
//   K     272  : fc? 0.5 * W_emb_ar @ W_ih[n,:64] : 0                 (AR feedback Q)
//   K 273..287 : 0
// (statics + all biases are folded into the accumulator-init sp[m,n])
__device__ __forceinline__ float wprime(
    int K, int n, bool fc,
    const float* W_hh, const float* W_ih,
    const float* W_emb_hc, const float* W_emb_fc, const float* W_emb_ar)
{
  if (K < HID) return W_hh[(size_t)n * HID + K];
  const float* wi = W_ih + (size_t)n * 96;
  if (K < 272) {
    const int d = K - 256;
    const float* we = fc ? W_emb_fc : W_emb_hc;
    float s = 0.f;
    for (int e = 0; e < EIN; ++e) s += we[(size_t)d * EIN + e] * wi[e];
    return fc ? 0.5f * s : s;
  }
  if (K == 272) {
    if (!fc) return 0.f;
    float s = 0.f;
    for (int e = 0; e < EIN; ++e) s += W_emb_ar[e] * wi[e];
    return 0.5f * s;
  }
  return 0.f;
}

// Build one 32x16 f16 B-fragment (K-tile base ktb, column n).
// Element j of lane (r,hi) holds K = ktb + j + (j>=8 ? 8 : 0) + 8*hi.
#define BUILD_FRAG(DST, KTB, N, FC)                                             \
  do {                                                                          \
    v16h _f;                                                                    \
    _Pragma("unroll")                                                           \
    for (int _j = 0; _j < 16; ++_j) {                                           \
      const int _K = (KTB) + _j + ((_j >= 8) ? 8 : 0) + 8 * hi;                 \
      _f[_j] = (_Float16)wprime(_K, (N), (FC), W_hh, W_ih,                      \
                                W_emb_hc, W_emb_fc, W_emb_ar);                  \
    }                                                                           \
    DST = _f;                                                                   \
  } while (0)

__global__ __launch_bounds__(512, 1)
void arlstm_persistent(const float* __restrict__ x_d_hc, const float* __restrict__ x_d_fc,
                       const float* __restrict__ x_s,
                       const float* __restrict__ W_emb_hc, const float* __restrict__ b_emb_hc,
                       const float* __restrict__ W_emb_fc, const float* __restrict__ b_emb_fc,
                       const float* __restrict__ W_emb_ar, const float* __restrict__ b_emb_ar,
                       const float* __restrict__ W_ih,     const float* __restrict__ W_hh,
                       const float* __restrict__ b_ih,     const float* __restrict__ b_hh,
                       const float* __restrict__ W_out,    const float* __restrict__ b_out,
                       float* __restrict__ y, int T_hc, int T_fc)
{
  extern __shared__ char smem[];
  _Float16* hbuf = (_Float16*)smem;           // 2 x [BM][HSTR] state rows (f16), ping-pong

  const int tid = threadIdx.x;
  const int w   = tid >> 5;       // wave 0..15 -> hidden group
  const int l   = tid & 31;
  const int r   = l & 15;
  const int hi  = l >> 4;
  const int wgbase = blockIdx.x * BM;

  // per-wave LDS block: 16 weight fragments + sp, all reachable via imm offsets
  char* const wvl = smem + WLDS_OFF + w * WBLK + l * 32;   // frag(slot,nt) = wvl + (slot*4+nt)*1024
  char* const spl = smem + WLDS_OFF + w * WBLK + SPV_OFF + l * 16;  // sp(nt) = spl + nt*512

  // ---------------- acc-init sp fill (thread-private LDS slot, no barriers needed) ----
  // sp[m,n] = sum_s x_s[m,s]*W_ih[n,64+s] + b_ih[n]+b_hh[n] + (phase embedding-bias term)
  #define FILL_SP(FC)                                                            \
    do {                                                                         \
      _Pragma("unroll")                                                          \
      for (int nt = 0; nt < 4; ++nt) {                                           \
        const int n = nt * HID + w * 16 + r;                                     \
        const float* wi = W_ih + (size_t)n * 96;                                 \
        float bias = b_ih[n] + b_hh[n];                                          \
        float d = 0.f;                                                           \
        if (FC) { for (int e = 0; e < EIN; ++e) d += (b_emb_fc[e] + b_emb_ar[e]) * wi[e]; bias += 0.5f * d; } \
        else    { for (int e = 0; e < EIN; ++e) d += b_emb_hc[e] * wi[e];                 bias += d; }         \
        v8h sv;                                                                  \
        _Pragma("unroll")                                                        \
        for (int i = 0; i < 8; ++i) {                                            \
          const int m = wgbase + i + 8 * hi;                                     \
          float s = bias;                                                        \
          for (int s2 = 0; s2 < SIN; ++s2) s += x_s[(size_t)m * SIN + s2] * wi[64 + s2]; \
          sv[i] = (_Float16)s;                                                   \
        }                                                                        \
        *(v8h*)(spl + nt * 512) = sv;                                            \
      }                                                                          \
    } while (0)

  FILL_SP(false);

  // ---------------- init both state buffers ----------------
  for (int i = tid; i < 2 * BM * HSTR; i += blockDim.x) hbuf[i] = (_Float16)0.f;
  __syncthreads();
  if (tid < BM * DIN) {                                     // x_0 into buffer 0
    const int rr = tid >> 4, d = tid & 15;
    hbuf[rr * HSTR + 256 + d] =
        (_Float16)x_d_hc[((size_t)(wgbase + rr) * T_hc + 0) * DIN + d];
  }

  // ---------------- LDS-resident weight fragments: K-tiles 0..3 (pure W_hh) ----------
  #pragma unroll
  for (int slot = 0; slot < LDSKT; ++slot) {
    #pragma unroll
    for (int nt = 0; nt < 4; ++nt) {
      v16h f;
      BUILD_FRAG(f, slot * 32, nt * HID + w * 16 + r, false);
      *(v16h*)(wvl + (slot * 4 + nt) * WFRAG) = f;
    }
  }

  // ---------------- register-resident weight fragments: K-tiles 4..8 -----------------
  v16h wreg[5][4];
  #pragma unroll
  for (int kk = 0; kk < 5; ++kk)
    #pragma unroll
    for (int nt = 0; nt < 4; ++nt)
      BUILD_FRAG(wreg[kk][nt], (4 + kk) * 32, nt * HID + w * 16 + r, false);

  const float b0 = b_out[0];
  const int hid = w * 16 + r;     // hidden unit owned by this lane (elementwise)
  // ping-pong base pointers (buffer 0)
  const char* const abase0 = (const char*)(hbuf + r * HSTR + hi * 8);  // A-frag base
  char* const hsbase0 = (char*)(hbuf + 8 * hi * HSTR + hid);           // h-store base

  float cst[8];
  #pragma unroll
  for (int i = 0; i < 8; ++i) cst[i] = 0.f;

  __syncthreads();

  // =================== time loop ===================
  const int Ttot = T_hc + T_fc;
  for (int gt = 0; gt < Ttot; ++gt) {
    if (gt == T_hc) {     // phase switch: rebuild only phase K-tile 8 + refresh sp bias
      #pragma unroll
      for (int nt = 0; nt < 4; ++nt) BUILD_FRAG(wreg[4][nt], 8 * 32, nt * HID + w * 16 + r, true);
      FILL_SP(true);      // sp slots are thread-private -> no barrier needed
    }

    const int co = (gt & 1) ? HBUF_BYTES : 0;     // read buffer offset
    const int no = HBUF_BYTES - co;               // write buffer offset
    const v8h* const ab = (const v8h*)(abase0 + co);

    // ---- acc init from sp (statics + biases) ----
    v8f acc[4];
    #pragma unroll
    for (int nt = 0; nt < 4; ++nt) {
      const v8h sv = *(const v8h*)(spl + nt * 512);
      #pragma unroll
      for (int i = 0; i < 8; ++i) acc[nt][i] = (float)sv[i];
    }

    // ---- gates += state_ext @ W' : 9 K-tiles x 4 matched N-tiles ----
    #pragma unroll
    for (int ks = 0; ks < LDSKT; ++ks) {        // LDS weight K-tiles 0..3
      AFrag a;
      a.p.lo = ab[ks * 4];
      a.p.hi = ab[ks * 4 + 2];
      #pragma unroll
      for (int nt = 0; nt < 4; ++nt) {
        const v16h wf = *(const v16h*)(wvl + (ks * 4 + nt) * WFRAG);
        acc[nt] = __builtin_amdgcn_wmma_f32_16x16x32_f16(false, a.v, false, wf, (short)0, acc[nt], false, false);
      }
    }
    #pragma unroll
    for (int kk = 0; kk < 5; ++kk) {            // register weight K-tiles 4..8
      AFrag a;
      a.p.lo = ab[(4 + kk) * 4];
      a.p.hi = ab[(4 + kk) * 4 + 2];
      #pragma unroll
      for (int nt = 0; nt < 4; ++nt)
        acc[nt] = __builtin_amdgcn_wmma_f32_16x16x32_f16(false, a.v, false, wreg[kk][nt], (short)0, acc[nt], false, false);
    }

    // ---- LSTM cell: i=acc[0], f=acc[1], g=acc[2], o=acc[3]; write into OTHER buffer ----
    {
      _Float16* const hstore = (_Float16*)(hsbase0 + no);
      #pragma unroll
      for (int i = 0; i < 8; ++i) {
        const float c = sigmoidf_(acc[1][i]) * cst[i] +
                        sigmoidf_(acc[0][i]) * tanhf_fast(acc[2][i]);
        cst[i] = c;
        const float h = sigmoidf_(acc[3][i]) * tanhf_fast(c);
        hstore[i * HSTR] = (_Float16)h;
      }
    }

    // ---- stage next timestep's dynamic input into the write buffer ----
    const int nxt = gt + 1;
    if (nxt < Ttot && tid < BM * DIN) {
      const int rr = tid >> 4, d = tid & 15;
      const float xv = (nxt < T_hc)
          ? x_d_hc[((size_t)(wgbase + rr) * T_hc + nxt) * DIN + d]
          : x_d_fc[((size_t)(wgbase + rr) * T_fc + (nxt - T_hc)) * DIN + d];
      *(_Float16*)(smem + no + (rr * HSTR + 256 + d) * 2) = (_Float16)xv;
    }
    __syncthreads();   // write buffer complete; it becomes next step's read buffer

    // ---- prediction head Q = h @ W_out + b_out (wave 0), feeds AR input + output ----
    if (gt + 1 >= T_hc) {
      if (w == 0) {
        const int m = l >> 1, part = l & 1;
        float s = 0.f;
        const _Float16* hp = (const _Float16*)(smem + no + (m * HSTR + part * 128) * 2);
        const float*    wo = W_out + part * 128;
        for (int k = 0; k < 128; ++k) s += (float)hp[k] * wo[k];
        s += __shfl_xor(s, 1, 32);
        if (part == 0) {
          const float q = s + b0;
          *(_Float16*)(smem + no + (m * HSTR + 272) * 2) = (_Float16)q;  // AR feedback
          if (gt >= T_hc) y[(size_t)(wgbase + m) * T_fc + (gt - T_hc)] = q;
        }
      }
      __syncthreads();
    }
  }
}

extern "C" void kernel_launch(void* const* d_in, const int* in_sizes, int n_in,
                              void* d_out, int out_size, void* d_ws, size_t ws_size,
                              hipStream_t stream) {
  (void)n_in; (void)out_size; (void)d_ws; (void)ws_size;
  const float* x_d_hc   = (const float*)d_in[0];
  const float* x_d_fc   = (const float*)d_in[1];
  const float* x_s      = (const float*)d_in[2];
  const float* W_emb_hc = (const float*)d_in[3];
  const float* b_emb_hc = (const float*)d_in[4];
  const float* W_emb_fc = (const float*)d_in[5];
  const float* b_emb_fc = (const float*)d_in[6];
  const float* W_emb_ar = (const float*)d_in[7];
  const float* b_emb_ar = (const float*)d_in[8];
  const float* W_ih     = (const float*)d_in[9];
  const float* W_hh     = (const float*)d_in[10];
  const float* b_ih     = (const float*)d_in[11];
  const float* b_hh     = (const float*)d_in[12];
  const float* W_out    = (const float*)d_in[13];
  const float* b_out    = (const float*)d_in[14];
  float* y = (float*)d_out;

  const int B    = in_sizes[2] / SIN;          // 1024
  const int T_hc = in_sizes[0] / (B * DIN);    // 365
  const int T_fc = in_sizes[1] / (B * DIN);    // 48

  hipFuncSetAttribute((const void*)arlstm_persistent,
                      hipFuncAttributeMaxDynamicSharedMemorySize, LDS_BYTES);

  dim3 grid(B / BM);   // 64 workgroups, each owns 16 batch rows for all timesteps
  dim3 block(NW * 32); // 16 waves; each owns matched i/f/g/o tiles of one hidden group
  arlstm_persistent<<<grid, block, LDS_BYTES, stream>>>(
      x_d_hc, x_d_fc, x_s, W_emb_hc, b_emb_hc, W_emb_fc, b_emb_fc,
      W_emb_ar, b_emb_ar, W_ih, W_hh, b_ih, b_hh, W_out, b_out, y, T_hc, T_fc);
}
